// MultiHeadAttention_48026324304093
// MI455X (gfx1250) — matrix operand order repaired and checked by reading, not compiled, not run on hardware
//
#include <hip/hip_runtime.h>
#include <hip/hip_bf16.h>

// ---------------------------------------------------------------------------
// MHA for B=2, N=2048, EMB=1024, H=16 (d=64) on gfx1250 (wave32, WMMA).
// bf16 inputs / f32 accumulate via v_wmma_f32_16x16x32_bf16.
// LDS double-buffered staging via GLOBAL_LOAD_ASYNC_TO_LDS_B128 (ASYNCcnt).
// ---------------------------------------------------------------------------

typedef __attribute__((ext_vector_type(16))) __bf16 v16bf;
typedef __attribute__((ext_vector_type(8)))  float  v8f;
typedef __attribute__((ext_vector_type(4)))  int    v4i;

union Frag16 {                   // 16 bf16 = one WMMA A or B operand (8 VGPRs)
    v16bf v;
    uint4 q[2];
    unsigned short u[16];
};

#if __has_builtin(__builtin_amdgcn_global_load_async_to_lds_b128)
#define HAVE_ASYNC_LDS 1
#else
#define HAVE_ASYNC_LDS 0
#endif

typedef __attribute__((address_space(1))) v4i* gv4i_p;   // global int32x4*
typedef __attribute__((address_space(3))) v4i* lv4i_p;   // LDS    int32x4*

// copy 16 bytes global -> LDS (async if available)
__device__ __forceinline__ void cp16(unsigned short* l, const unsigned short* g) {
#if HAVE_ASYNC_LDS
    __builtin_amdgcn_global_load_async_to_lds_b128(
        (gv4i_p)g, (lv4i_p)l, /*offset=*/0, /*cpol=*/0);
#else
    *(uint4*)l = *(const uint4*)g;
#endif
}

__device__ __forceinline__ void async_join() {
#if HAVE_ASYNC_LDS
#if __has_builtin(__builtin_amdgcn_s_wait_asynccnt)
    __builtin_amdgcn_s_wait_asynccnt(0);
#else
    asm volatile("s_wait_asynccnt 0" ::: "memory");
#endif
#endif
}

__device__ __forceinline__ unsigned short f2bf(float f) {
    unsigned int u = __float_as_uint(f);
    unsigned int r = (u + 0x7FFFu + ((u >> 16) & 1u)) >> 16;   // RNE
    return (unsigned short)r;
}

__device__ __forceinline__ v8f wmma_bf16(const Frag16& a, const Frag16& b, v8f c) {
    return __builtin_amdgcn_wmma_f32_16x16x32_bf16(
        /*neg_a=*/false, a.v, /*neg_b=*/false, b.v,
        /*c_mod=*/(short)0, c, /*reuse_a=*/false, /*reuse_b=*/false);
}

// ----------------------------- converters ---------------------------------

__global__ __launch_bounds__(256) void f32_to_bf16_kernel(
    const float* __restrict__ in, unsigned short* __restrict__ out, int n) {
    int i = blockIdx.x * 256 + threadIdx.x;
    if (i < n) out[i] = f2bf(in[i]);
}

// W [K,N] f32 row-major  ->  Bt [N,K] bf16 (so WMMA B-fragments are contiguous)
__global__ __launch_bounds__(256) void transpose_to_bf16_kernel(
    const float* __restrict__ W, unsigned short* __restrict__ Bt, int K, int N) {
    int i = blockIdx.x * 256 + threadIdx.x;     // i = j*K + k (coalesced writes)
    if (i < N * K) {
        int j = i / K, k = i - j * K;
        Bt[i] = f2bf(W[(size_t)k * N + j]);
    }
}

// ----------------------------- bf16 GEMM -----------------------------------
// C = A[M,K](bf16) * Bt[N,K]^T(bf16) + bias.
// Block: 256 thr = 8 waves (2 along M x 4 along N). WG tile 64x256, k-step 32.
// Double-buffered LDS staging: sA 2x4KB, sB 2x16KB (40KB of 320KB WGP LDS).
// mode 0: out f32 [M,N]
// mode 1: out bf16, split heads       [B,H,Nseq,64] (Q, K)
// mode 2: out bf16, head-transposed   [B,H,64,Nseq] (V)
__global__ __launch_bounds__(256) void gemm_bf16_kernel(
    const unsigned short* __restrict__ A, const unsigned short* __restrict__ Bt,
    const float* __restrict__ bias, void* __restrict__ outp,
    int M, int N, int K, int Nseq, int mode) {

    __shared__ __align__(16) unsigned short sA[2][64][32];
    __shared__ __align__(16) unsigned short sB[2][256][32];

    const int tid  = threadIdx.x;
    const int lane = tid & 31;
    const int wave = tid >> 5;
    const int l15  = lane & 15;
    const int half = lane >> 4;
    const int mwave = wave & 1;
    const int nwave = wave >> 1;
    const int rowBase0 = blockIdx.y * 64;
    const int colBase0 = blockIdx.x * 256;

    // staging assignment: A tile 64x32 = 256 x 16B chunks (1/thread),
    //                     B tile 256x32 = 1024 x 16B chunks (4/thread)
    const int arow = tid >> 2, aoff = (tid & 3) * 8;
    const unsigned short* agp = A + (size_t)(rowBase0 + arow) * K + aoff;
    const unsigned short* bgp = Bt + (size_t)(colBase0 + tid) * K;

    v8f acc[2][4];
    for (int mt = 0; mt < 2; ++mt)
        for (int nt = 0; nt < 4; ++nt)
            for (int r = 0; r < 8; ++r) acc[mt][nt][r] = 0.0f;

    const int nk = K / 32;

    // prologue: stage k-slice 0 into buffer 0
    {
        cp16(&sA[0][arow][aoff], agp);
        for (int c = 0; c < 4; ++c) cp16(&sB[0][tid][c * 8], bgp + c * 8);
    }

    for (int i = 0; i < nk; ++i) {
        const int cur = i & 1;
        async_join();           // my async writes into buf[cur] have landed
        __syncthreads();        // everyone's landed; buf[cur^1] fully consumed

        if (i + 1 < nk) {       // stage next k-slice while computing this one
            const int kk = (i + 1) * 32;
            cp16(&sA[cur ^ 1][arow][aoff], agp + kk);
            for (int c = 0; c < 4; ++c)
                cp16(&sB[cur ^ 1][tid][c * 8], bgp + kk + c * 8);
        }

        Frag16 a[2], b[4];
        for (int mt = 0; mt < 2; ++mt) {
            const unsigned short* ap = &sA[cur][mwave * 32 + mt * 16 + l15][0];
            a[mt].q[0] = *(const uint4*)(ap + 8 * half);        // K = 8h..8h+7
            a[mt].q[1] = *(const uint4*)(ap + 16 + 8 * half);   // K = 16+8h..
        }
        for (int nt = 0; nt < 4; ++nt) {
            const unsigned short* bp =
                &sB[cur][nwave * 64 + nt * 16 + l15][0];
            b[nt].q[0] = *(const uint4*)(bp + 8 * half);        // K = 8h..8h+7
            b[nt].q[1] = *(const uint4*)(bp + 16 + 8 * half);   // K = 16+8h..
        }
        for (int mt = 0; mt < 2; ++mt)
            for (int nt = 0; nt < 4; ++nt)
                acc[mt][nt] = wmma_bf16(a[mt], b[nt], acc[mt][nt]);
    }

    // epilogue: C element (m = r + 8*half, n = l15) per 16x16 tile
    for (int nt = 0; nt < 4; ++nt) {
        const int col = colBase0 + nwave * 64 + nt * 16 + l15;
        const float bv = bias[col];
        for (int mt = 0; mt < 2; ++mt) {
            for (int r = 0; r < 8; ++r) {
                const int row = rowBase0 + mwave * 32 + mt * 16 + r + 8 * half;
                const float val = acc[mt][nt][r] + bv;
                if (mode == 0) {
                    ((float*)outp)[(size_t)row * N + col] = val;
                } else {
                    const int bb = row / Nseq, nr = row - bb * Nseq;
                    const int h = col >> 6, dd = col & 63;
                    size_t dst;
                    if (mode == 1)
                        dst = ((size_t)(bb * 16 + h) * Nseq + nr) * 64 + dd;
                    else
                        dst = ((size_t)(bb * 16 + h) * 64 + dd) * Nseq + nr;
                    ((unsigned short*)outp)[dst] = f2bf(val);
                }
            }
        }
    }
}

// -------------------------- flash attention --------------------------------
// grid: (Nseq/64, B*H), block 128 (4 waves; wave = 16 query rows).
// Q,K: [B,H,Nseq,64] bf16; Vt: [B,H,64,Nseq] bf16; out: [B,Nseq,EMB] bf16.
// K/V tiles for each 32-key block are shared by all 4 waves -> stage in LDS
// (double-buffered 2x(4KB+4KB)) with async copies.
__global__ __launch_bounds__(128) void attn_kernel(
    const unsigned short* __restrict__ Qm, const unsigned short* __restrict__ Km,
    const unsigned short* __restrict__ Vt, unsigned short* __restrict__ Ao,
    int Nseq) {

    __shared__ __align__(16) unsigned short sK[2][32][64];   // [key][d]
    __shared__ __align__(16) unsigned short sV[2][64][32];   // [d][key]
    __shared__ __align__(16) unsigned short ldsP[4][16][32];

    const int tid  = threadIdx.x;
    const int lane = tid & 31;
    const int wave = tid >> 5;
    const int l15  = lane & 15;
    const int half = lane >> 4;
    const int bh = blockIdx.y;                       // b*16 + h
    const int q0 = blockIdx.x * 64 + wave * 16;

    const unsigned short* Qb = Qm + (size_t)bh * Nseq * 64;
    const unsigned short* Kb = Km + (size_t)bh * Nseq * 64;
    const unsigned short* Vb = Vt + (size_t)bh * 64 * Nseq;

    // Q A-fragments for the two K-chunks of d=64; reused across all key blocks
    Frag16 aq[2];
    {
        const unsigned short* qp = Qb + (size_t)(q0 + l15) * 64;
        for (int kc = 0; kc < 2; ++kc) {
            aq[kc].q[0] = *(const uint4*)(qp + kc * 32 + 8 * half);
            aq[kc].q[1] = *(const uint4*)(qp + kc * 32 + 16 + 8 * half);
        }
    }

    v8f o[4];
    for (int t = 0; t < 4; ++t)
        for (int r = 0; r < 8; ++r) o[t][r] = 0.0f;
    float mi[8], li[8];
    for (int r = 0; r < 8; ++r) { mi[r] = -1e30f; li[r] = 0.0f; }

    const float scale = 0.03125f;                    // 1/sqrt(EMB) = 1/32

    // staging: K tile 32x64 = 256 chunks, V tile 64x32 = 256 chunks; 128 thr
    // -> 2 chunks each per tile per thread.
    const int nkb = Nseq / 32;

    {   // prologue: stage key-block 0 into buffer 0
        for (int s = 0; s < 2; ++s) {
            const int c = tid + s * 128;
            cp16(&sK[0][c >> 3][(c & 7) * 8],
                 Kb + (size_t)(c >> 3) * 64 + (c & 7) * 8);
            cp16(&sV[0][c >> 2][(c & 3) * 8],
                 Vb + (size_t)(c >> 2) * Nseq + (c & 3) * 8);
        }
    }

    for (int i = 0; i < nkb; ++i) {
        const int kb = i * 32;
        const int cur = i & 1;
        async_join();
        __syncthreads();

        if (i + 1 < nkb) {
            const int kb2 = kb + 32;
            for (int s = 0; s < 2; ++s) {
                const int c = tid + s * 128;
                cp16(&sK[cur ^ 1][c >> 3][(c & 7) * 8],
                     Kb + (size_t)(kb2 + (c >> 3)) * 64 + (c & 7) * 8);
                cp16(&sV[cur ^ 1][c >> 2][(c & 3) * 8],
                     Vb + (size_t)(c >> 2) * Nseq + kb2 + (c & 3) * 8);
            }
        }

        // ---- S = Q * K^T for 16 queries x 32 keys (4 WMMAs) ----
        v8f s[2];
        for (int f = 0; f < 2; ++f)
            for (int r = 0; r < 8; ++r) s[f][r] = 0.0f;
        for (int f = 0; f < 2; ++f) {
            for (int kc = 0; kc < 2; ++kc) {
                Frag16 bk;
                const unsigned short* kp =
                    &sK[cur][f * 16 + l15][kc * 32];
                bk.q[0] = *(const uint4*)(kp + 8 * half);
                bk.q[1] = *(const uint4*)(kp + 16 + 8 * half);
                s[f] = wmma_bf16(aq[kc], bk, s[f]);
            }
        }

        // ---- online softmax (rows m = r + 8*half; cols across 16 lanes) ----
        for (int r = 0; r < 8; ++r) {
            float v0 = s[0][r] * scale;
            float v1 = s[1][r] * scale;
            float mx = fmaxf(v0, v1);
            mx = fmaxf(mx, __shfl_xor(mx, 1, 32));
            mx = fmaxf(mx, __shfl_xor(mx, 2, 32));
            mx = fmaxf(mx, __shfl_xor(mx, 4, 32));
            mx = fmaxf(mx, __shfl_xor(mx, 8, 32));
            const float mnew = fmaxf(mi[r], mx);
            const float e0 = __expf(v0 - mnew);
            const float e1 = __expf(v1 - mnew);
            float rs = e0 + e1;
            rs += __shfl_xor(rs, 1, 32);
            rs += __shfl_xor(rs, 2, 32);
            rs += __shfl_xor(rs, 4, 32);
            rs += __shfl_xor(rs, 8, 32);
            const float alpha = __expf(mi[r] - mnew);
            li[r] = li[r] * alpha + rs;
            mi[r] = mnew;
            for (int t = 0; t < 4; ++t) o[t][r] *= alpha;
            // stage P (C-frag layout) into LDS for re-layout as an A-fragment
            ldsP[wave][r + 8 * half][l15]      = f2bf(e0);
            ldsP[wave][r + 8 * half][16 + l15] = f2bf(e1);
        }

        // ---- P as A-fragment (16x32) from LDS (same-wave DS ops in-order) ----
        Frag16 pf;
        pf.q[0] = *(const uint4*)&ldsP[wave][l15][8 * half];
        pf.q[1] = *(const uint4*)&ldsP[wave][l15][16 + 8 * half];

        // ---- O += P * V  (V tile in LDS, [d][key]: contiguous B-frags) ----
        for (int t = 0; t < 4; ++t) {
            Frag16 bv;
            const unsigned short* vp = &sV[cur][t * 16 + l15][0];
            bv.q[0] = *(const uint4*)(vp + 8 * half);
            bv.q[1] = *(const uint4*)(vp + 16 + 8 * half);
            o[t] = wmma_bf16(pf, bv, o[t]);
        }
    }

    // ---- normalize and write merged-head bf16 [B, Nseq, EMB] ----
    const int bb = bh >> 4, h = bh & 15;
    for (int r = 0; r < 8; ++r) {
        const float inv = 1.0f / li[r];
        const size_t row = (size_t)bb * Nseq + q0 + r + 8 * half;
        for (int t = 0; t < 4; ++t) {
            Ao[row * 1024 + h * 64 + t * 16 + l15] = f2bf(o[t][r] * inv);
        }
    }
}

// ----------------------------- launcher ------------------------------------

extern "C" void kernel_launch(void* const* d_in, const int* in_sizes, int n_in,
                              void* d_out, int out_size, void* d_ws, size_t ws_size,
                              hipStream_t stream) {
    (void)in_sizes; (void)n_in; (void)out_size; (void)ws_size;

    const float* x  = (const float*)d_in[0];
    const float* Wq = (const float*)d_in[1];
    const float* bq = (const float*)d_in[2];
    const float* Wk = (const float*)d_in[3];
    const float* bk = (const float*)d_in[4];
    const float* Wv = (const float*)d_in[5];
    const float* bv = (const float*)d_in[6];
    const float* Wo = (const float*)d_in[7];
    const float* bo = (const float*)d_in[8];

    const int EMB = 1024, Bsz = 2, Nseq = 2048;
    const int Mrows = Bsz * Nseq;                       // 4096
    const size_t MB = 1ull << 20;

    char* ws = (char*)d_ws;
    unsigned short* xbf = (unsigned short*)(ws + 0 * MB);   // 8 MB
    unsigned short* wqt = (unsigned short*)(ws + 8 * MB);   // 2 MB each
    unsigned short* wkt = (unsigned short*)(ws + 10 * MB);
    unsigned short* wvt = (unsigned short*)(ws + 12 * MB);
    unsigned short* wot = (unsigned short*)(ws + 14 * MB);
    unsigned short* Qb  = (unsigned short*)(ws + 16 * MB);  // 8 MB
    unsigned short* Kb  = (unsigned short*)(ws + 24 * MB);  // 8 MB
    unsigned short* Vtb = (unsigned short*)(ws + 32 * MB);  // 8 MB
    unsigned short* att = (unsigned short*)(ws + 40 * MB);  // 8 MB

    // 1) dtype conversion / weight transposition
    f32_to_bf16_kernel<<<(Mrows * EMB) / 256, 256, 0, stream>>>(x, xbf, Mrows * EMB);
    transpose_to_bf16_kernel<<<(EMB * EMB) / 256, 256, 0, stream>>>(Wq, wqt, EMB, EMB);
    transpose_to_bf16_kernel<<<(EMB * EMB) / 256, 256, 0, stream>>>(Wk, wkt, EMB, EMB);
    transpose_to_bf16_kernel<<<(EMB * EMB) / 256, 256, 0, stream>>>(Wv, wvt, EMB, EMB);
    transpose_to_bf16_kernel<<<(EMB * EMB) / 256, 256, 0, stream>>>(Wo, wot, EMB, EMB);

    // 2) QKV projections (WMMA GEMMs)
    dim3 ggrid(EMB / 256, Mrows / 64);                  // (4, 64)
    gemm_bf16_kernel<<<ggrid, 256, 0, stream>>>(xbf, wqt, bq, Qb,  Mrows, EMB, EMB, Nseq, 1);
    gemm_bf16_kernel<<<ggrid, 256, 0, stream>>>(xbf, wkt, bk, Kb,  Mrows, EMB, EMB, Nseq, 1);
    gemm_bf16_kernel<<<ggrid, 256, 0, stream>>>(xbf, wvt, bv, Vtb, Mrows, EMB, EMB, Nseq, 2);

    // 3) flash attention (WMMA)
    attn_kernel<<<dim3(Nseq / 64, Bsz * 16), 128, 0, stream>>>(Qb, Kb, Vtb, att, Nseq);

    // 4) output projection -> fp32 d_out
    gemm_bf16_kernel<<<ggrid, 256, 0, stream>>>(att, wot, bo, (float*)d_out,
                                                Mrows, EMB, EMB, Nseq, 0);
}
